// DSQGBlock_75505525063708
// MI455X (gfx1250) — compile-verified, matrix operand order checked by reading, not run on hardware
//
#include <hip/hip_runtime.h>

// ---------------------------------------------------------------------------
// Problem constants (B=1)
// ---------------------------------------------------------------------------
#define NT   2048       // tokens
#define DIM  1024       // model dim
#define NH   16         // heads
#define HD   64         // head dim
#define FFN  2816       // ffn dim
#define NOFF 44         // number of offsets
#define QKVN (3*DIM)    // 3072

typedef __attribute__((ext_vector_type(16))) __bf16 bf16x16;
typedef __attribute__((ext_vector_type(2)))  __bf16 bf16x2;
typedef __attribute__((ext_vector_type(8)))  float  f32x8;
typedef __attribute__((ext_vector_type(2)))  float  f32x2;

__device__ __forceinline__ bf16x2 pack_bf16(float a, float b) {
    f32x2 f;
    f.x = a;
    f.y = b;
    return __builtin_convertvector(f, bf16x2);   // -> v_cvt_pk_bf16_f32
}

// ---------------------------------------------------------------------------
// RMSNorm: one block per row, fp32 in -> bf16 out (GEMM A operand)
// ---------------------------------------------------------------------------
__global__ __launch_bounds__(256)
void rmsnorm_k(const float* __restrict__ x, const float* __restrict__ scale,
               __bf16* __restrict__ out) {
    const int row = blockIdx.x;
    const int t   = threadIdx.x;
    __shared__ float red[256];
    const float* xr = x + (size_t)row * DIM;
    float ss = 0.f;
    for (int i = t; i < DIM; i += 256) { float v = xr[i]; ss += v * v; }
    red[t] = ss;
    __syncthreads();
    for (int s = 128; s > 0; s >>= 1) {
        if (t < s) red[t] += red[t + s];
        __syncthreads();
    }
    const float inv = 1.0f / sqrtf(red[0] * (1.0f / DIM) + 1e-6f);
    for (int i = t * 2; i < DIM; i += 512) {
        bf16x2 p = pack_bf16(scale[i] * xr[i] * inv,
                             scale[i + 1] * xr[i + 1] * inv);
        *(bf16x2*)&out[(size_t)row * DIM + i] = p;
    }
}

// ---------------------------------------------------------------------------
// bf16 WMMA GEMM:  C[M,N] = A_bf16[M,K] @ Bw_f32[K,N]   (fp32 accumulate)
// Compile-time M,N,K. Block tile 128x128, K-step 32, 256 thr = 8 waves (2x4),
// wave owns 64x32 (8 WMMA accumulators). LDS double-buffered; next K-tile's
// global loads prefetched into registers before computing the current tile.
// Weights converted fp32->bf16 (v_cvt_pk_bf16_f32) into LDS transposed [n][k].
// Epilogues:
//   0: outF = acc
//   1: outF = sigmoid(acc + bias[n])
//   2: outF = silu(acc)
//   3: outB = bf16(acc * aux[m,n])
//   4: outF = acc + aux[m,n]
// ---------------------------------------------------------------------------

struct StageRegs {
    uint4 a0, a1;      // 8+8 bf16 of A row segment
    float b[16];       // [0..7] = row k, [8..15] = row k+1  (8 columns)
};

template <int N, int K>
__device__ __forceinline__ void stage_load(const __bf16* __restrict__ A,
                                           const float* __restrict__ Bw,
                                           int m0, int n0, int kt,
                                           int ar, int ac, int bk2, int bn8,
                                           StageRegs& r) {
    const uint4* ag = (const uint4*)(A + (size_t)(m0 + ar) * K + kt * 32 + ac);
    r.a0 = ag[0];
    r.a1 = ag[1];
    const float* bg0 = Bw + (size_t)(kt * 32 + bk2) * N + n0 + bn8;
    const float* bg1 = bg0 + N;
    #pragma unroll
    for (int j = 0; j < 2; ++j) {
        ((float4*)r.b)[j]     = ((const float4*)bg0)[j];
        ((float4*)r.b)[2 + j] = ((const float4*)bg1)[j];
    }
}

__device__ __forceinline__ void stage_store(__bf16* __restrict__ As,
                                            __bf16* __restrict__ Bs,
                                            int ar, int ac, int bk2, int bn8,
                                            const StageRegs& r) {
    *(uint4*)&As[ar * 32 + ac]     = r.a0;
    *(uint4*)&As[ar * 32 + ac + 8] = r.a1;
    #pragma unroll
    for (int j = 0; j < 8; ++j) {
        // pack (k, k+1) for column bn8+j -> one dword store, Bs[n][k]
        bf16x2 p = pack_bf16(r.b[j], r.b[8 + j]);
        *(bf16x2*)&Bs[(bn8 + j) * 32 + bk2] = p;
    }
}

__device__ __forceinline__ void compute_step(const __bf16* __restrict__ As,
                                             const __bf16* __restrict__ Bs,
                                             int wm, int wn, int lm, int hh,
                                             f32x8 acc[4][2]) {
    bf16x16 afr[4], bfr[2];
    #pragma unroll
    for (int i = 0; i < 4; ++i) {
        const __bf16* p = &As[(wm + i * 16 + lm) * 32 + hh * 8];
        *(uint4*)&afr[i]       = *(const uint4*)p;          // K = hh*8 .. +7
        *((uint4*)&afr[i] + 1) = *(const uint4*)(p + 16);   // K = 16+hh*8 .. +7
    }
    #pragma unroll
    for (int j = 0; j < 2; ++j) {
        const __bf16* p = &Bs[(wn + j * 16 + lm) * 32 + hh * 8];
        *(uint4*)&bfr[j]       = *(const uint4*)p;
        *((uint4*)&bfr[j] + 1) = *(const uint4*)(p + 16);
    }
    #pragma unroll
    for (int i = 0; i < 4; ++i)
        #pragma unroll
        for (int j = 0; j < 2; ++j)
            acc[i][j] = __builtin_amdgcn_wmma_f32_16x16x32_bf16(
                false, afr[i], false, bfr[j], (short)0, acc[i][j],
                false, false);
}

template <int EPI, int M, int N, int K>
__global__ __launch_bounds__(256)
void gemm_bf16_k(const __bf16* __restrict__ A, const float* __restrict__ Bw,
                 float* __restrict__ outF, __bf16* __restrict__ outB,
                 const float* __restrict__ aux, const float* __restrict__ bias) {
    __shared__ __bf16 As0[128 * 32], As1[128 * 32];   // [m][k], 8 KB each
    __shared__ __bf16 Bs0[128 * 32], Bs1[128 * 32];   // [n][k] transposed

    const int t    = threadIdx.x;
    const int wave = t >> 5;
    const int lane = t & 31;
    const int lm   = lane & 15;        // row/col within 16x16 tile
    const int hh   = lane >> 4;        // lane-half -> K-octet / acc-row group
    const int wm   = (wave >> 2) * 64; // wave row offset in block tile
    const int wn   = (wave & 3) * 32;  // wave col offset in block tile
    const int m0   = blockIdx.y * 128;
    const int n0   = blockIdx.x * 128;

    f32x8 acc[4][2];
    #pragma unroll
    for (int i = 0; i < 4; ++i)
        #pragma unroll
        for (int j = 0; j < 2; ++j)
            #pragma unroll
            for (int r = 0; r < 8; ++r) acc[i][j][r] = 0.0f;

    // staging assignments
    const int ar  = t >> 1;            // A tile row 0..127
    const int ac  = (t & 1) * 16;      // A tile col 0 or 16
    const int bk2 = (t >> 4) * 2;      // B tile k-row pair 0,2,..,30
    const int bn8 = (t & 15) * 8;      // B tile n-col 0,8,..,120

    constexpr int KT = K / 32;         // even for K=1024 (32) and K=2816 (88)
    static_assert(KT % 2 == 0, "KT must be even");

    StageRegs rg;
    stage_load<N, K>(A, Bw, m0, n0, 0, ar, ac, bk2, bn8, rg);
    stage_store(As0, Bs0, ar, ac, bk2, bn8, rg);
    __syncthreads();

    for (int kt = 0; kt < KT; kt += 2) {
        // tile kt in buf0 ready; prefetch kt+1, compute kt, stage kt+1 -> buf1
        stage_load<N, K>(A, Bw, m0, n0, kt + 1, ar, ac, bk2, bn8, rg);
        compute_step(As0, Bs0, wm, wn, lm, hh, acc);
        stage_store(As1, Bs1, ar, ac, bk2, bn8, rg);
        __syncthreads();
        // tile kt+1 in buf1 ready; prefetch kt+2 (if any), compute, stage -> buf0
        if (kt + 2 < KT)
            stage_load<N, K>(A, Bw, m0, n0, kt + 2, ar, ac, bk2, bn8, rg);
        compute_step(As1, Bs1, wm, wn, lm, hh, acc);
        if (kt + 2 < KT)
            stage_store(As0, Bs0, ar, ac, bk2, bn8, rg);
        __syncthreads();
    }

    // --- epilogue: acc[i][j][r] -> C[m0+wm+i*16+hh*8+r][n0+wn+j*16+lm]
    #pragma unroll
    for (int i = 0; i < 4; ++i) {
        #pragma unroll
        for (int j = 0; j < 2; ++j) {
            const int col = n0 + wn + j * 16 + lm;
            #pragma unroll
            for (int r = 0; r < 8; ++r) {
                const int row = m0 + wm + i * 16 + hh * 8 + r;
                const size_t idx = (size_t)row * N + col;
                float v = acc[i][j][r];
                if constexpr (EPI == 0) {
                    outF[idx] = v;
                } else if constexpr (EPI == 1) {
                    v += bias[col];
                    outF[idx] = 1.0f / (1.0f + expf(-v));
                } else if constexpr (EPI == 2) {
                    outF[idx] = v / (1.0f + expf(-v));
                } else if constexpr (EPI == 3) {
                    outB[idx] = (__bf16)(v * aux[idx]);
                } else {  // EPI == 4
                    outF[idx] = v + aux[idx];
                }
            }
        }
    }
}

// ---------------------------------------------------------------------------
// Gather attention + sigmoid gate, one block per token.
// qkv row layout: [ q(1024) | k(1024) | v(1024) ]
// out: og_bf16[n][h*64+d] = (sum_o alpha * v) * gate[n][h*64+d]
// ---------------------------------------------------------------------------
__global__ __launch_bounds__(256)
void attn_k(const float* __restrict__ qkv, const float* __restrict__ gate,
            const float* __restrict__ pos_bias, const int* __restrict__ offsets,
            __bf16* __restrict__ og) {
    const int n = blockIdx.x;
    const int t = threadIdx.x;

    __shared__ float q_sh[DIM];
    __shared__ float sc[NH][48];      // scores -> alpha (padded)
    __shared__ int   off_sh[NOFF];

    const float* qrow = qkv + (size_t)n * QKVN;
    for (int i = t; i < DIM; i += 256) q_sh[i] = qrow[i];
    if (t < NOFF) off_sh[t] = offsets[t];
    __syncthreads();

    // scores
    for (int p = t; p < NH * NOFF; p += 256) {
        const int h = p / NOFF, o = p % NOFF;
        const int off = off_sh[o];
        float s;
        if (n < off) {
            s = -1e30f;
        } else {
            const int src = n - off;
            const float* kp = qkv + (size_t)src * QKVN + DIM + h * HD;
            const float* qh = q_sh + h * HD;
            float d = 0.f;
            for (int i = 0; i < HD; ++i) d += qh[i] * kp[i];
            s = d * 0.125f + pos_bias[o * NH + h];   // 1/sqrt(64)
        }
        sc[h][o] = s;
    }
    __syncthreads();

    // softmax over 44 candidates (one thread per head)
    if (t < NH) {
        const int h = t;
        float m = -1e30f;
        for (int o = 0; o < NOFF; ++o) m = fmaxf(m, sc[h][o]);
        float sum = 0.f;
        for (int o = 0; o < NOFF; ++o) {
            const float e = expf(sc[h][o] - m);
            sc[h][o] = e;
            sum += e;
        }
        const float inv = 1.0f / sum;
        for (int o = 0; o < NOFF; ++o) sc[h][o] *= inv;
    }
    __syncthreads();

    // weighted V sum (4 consecutive dims per thread), fused sigmoid-gate, bf16 out
    {
        const int p4 = t * 4;                    // 0..1020, within one head
        const int h  = p4 >> 6;
        const int d0 = p4 & 63;
        float a0 = 0.f, a1 = 0.f, a2 = 0.f, a3 = 0.f;
        for (int o = 0; o < NOFF; ++o) {
            int src = n - off_sh[o];
            if (src < 0) src = 0;                // alpha==0 there anyway
            const float* vp = qkv + (size_t)src * QKVN + 2 * DIM + h * HD + d0;
            const float w = sc[h][o];
            a0 += w * vp[0];
            a1 += w * vp[1];
            a2 += w * vp[2];
            a3 += w * vp[3];
        }
        const float* gp = gate + (size_t)n * DIM + p4;
        bf16x2 p01 = pack_bf16(a0 * gp[0], a1 * gp[1]);
        bf16x2 p23 = pack_bf16(a2 * gp[2], a3 * gp[3]);
        *(bf16x2*)&og[(size_t)n * DIM + p4]     = p01;
        *(bf16x2*)&og[(size_t)n * DIM + p4 + 2] = p23;
    }
}

// ---------------------------------------------------------------------------
// Launch
// ---------------------------------------------------------------------------
extern "C" void kernel_launch(void* const* d_in, const int* in_sizes, int n_in,
                              void* d_out, int out_size, void* d_ws, size_t ws_size,
                              hipStream_t stream) {
    const float* x        = (const float*)d_in[0];
    const float* n1s      = (const float*)d_in[1];
    const float* n2s      = (const float*)d_in[2];
    const float* w_qkv    = (const float*)d_in[3];
    const float* w_out    = (const float*)d_in[4];
    const float* w_gate   = (const float*)d_in[5];
    const float* b_gate   = (const float*)d_in[6];
    const float* pos_bias = (const float*)d_in[7];
    const float* w_fg     = (const float*)d_in[8];
    const float* w_fu     = (const float*)d_in[9];
    const float* w_fd     = (const float*)d_in[10];
    const int*   offs     = (const int*)d_in[11];
    float*       out      = (float*)d_out;

    char* w = (char*)d_ws;
    auto alloc = [&](size_t bytes) -> char* {
        char* p = w;
        w += (bytes + 255) & ~(size_t)255;
        return p;
    };
    __bf16* xn   = (__bf16*)alloc((size_t)NT * DIM * 2);
    __bf16* xn2  = (__bf16*)alloc((size_t)NT * DIM * 2);
    __bf16* og   = (__bf16*)alloc((size_t)NT * DIM * 2);
    __bf16* hbuf = (__bf16*)alloc((size_t)NT * FFN * 2);
    float*  qkv  = (float*)alloc((size_t)NT * QKVN * 4);
    float*  gate = (float*)alloc((size_t)NT * DIM * 4);
    float*  xmid = (float*)alloc((size_t)NT * DIM * 4);
    float*  sg   = (float*)alloc((size_t)NT * FFN * 4);

    const dim3 blk(256);
    const dim3 gN(NT);

    // 1) xn = rmsnorm(x) -> bf16
    rmsnorm_k<<<gN, blk, 0, stream>>>(x, n1s, xn);

    // 2) qkv = xn @ w_qkv  (fp32 out)
    gemm_bf16_k<0, NT, QKVN, DIM><<<dim3(QKVN / 128, NT / 128), blk, 0, stream>>>(
        xn, w_qkv, qkv, nullptr, nullptr, nullptr);

    // 3) gate = sigmoid(xn @ w_gate + b_gate)
    gemm_bf16_k<1, NT, DIM, DIM><<<dim3(DIM / 128, NT / 128), blk, 0, stream>>>(
        xn, w_gate, gate, nullptr, nullptr, b_gate);

    // 4) attention + gate fuse -> og bf16
    attn_k<<<gN, blk, 0, stream>>>(qkv, gate, pos_bias, offs, og);

    // 5) xmid = x + og @ w_out
    gemm_bf16_k<4, NT, DIM, DIM><<<dim3(DIM / 128, NT / 128), blk, 0, stream>>>(
        og, w_out, xmid, nullptr, x, nullptr);

    // 6) xn2 = rmsnorm(xmid) -> bf16
    rmsnorm_k<<<gN, blk, 0, stream>>>(xmid, n2s, xn2);

    // 7) sg = silu(xn2 @ w_ffn_gate)
    gemm_bf16_k<2, NT, FFN, DIM><<<dim3(FFN / 128, NT / 128), blk, 0, stream>>>(
        xn2, w_fg, sg, nullptr, nullptr, nullptr);

    // 8) h = (xn2 @ w_ffn_up) * sg   -> bf16
    gemm_bf16_k<3, NT, FFN, DIM><<<dim3(FFN / 128, NT / 128), blk, 0, stream>>>(
        xn2, w_fu, nullptr, hbuf, sg, nullptr);

    // 9) out = xmid + h @ w_ffn_down
    gemm_bf16_k<4, NT, DIM, FFN><<<dim3(DIM / 128, NT / 128), blk, 0, stream>>>(
        hbuf, w_fd, out, nullptr, xmid, nullptr);
}
